// AttentionHead_19026705121961
// MI455X (gfx1250) — matrix operand order repaired and checked by reading, not compiled, not run on hardware
//
#include <hip/hip_runtime.h>
#include <hip/hip_bf16.h>

typedef __attribute__((ext_vector_type(16))) _Float16 v16h;
typedef __attribute__((ext_vector_type(8)))  _Float16 v8h;
typedef __attribute__((ext_vector_type(8)))  float    v8f;
typedef __attribute__((ext_vector_type(4)))  float    v4f;
typedef __attribute__((ext_vector_type(4)))  unsigned int v4u;
typedef __attribute__((ext_vector_type(8)))  int      v8i;
typedef __attribute__((ext_vector_type(4)))  int      v4i;

#define B_ 512
#define S_ 256
#define E_ 384
#define H_ 64

static constexpr int NCOL      = 192;  // 3*H (Q|K|V columns)
static constexpr int WT_STRIDE = 40;   // 32 k + 8 pad halves: 80B rows, aligned + conflict-free
static constexpr int KS_STRIDE = 72;   // 64 h + 8 pad halves: 144B rows, aligned + conflict-free
static constexpr int PS_STRIDE = 40;   // 32 keys + 8 pad halves

#define HAVE_TDM __has_builtin(__builtin_amdgcn_tensor_load_to_lds)

__device__ __forceinline__ v8f vzero8() {
  v8f z;
#pragma unroll
  for (int i = 0; i < 8; ++i) z[i] = 0.0f;
  return z;
}

__device__ __forceinline__ v16h cat8(v8h lo, v8h hi) {
  v16h r;
#pragma unroll
  for (int i = 0; i < 8; ++i) { r[i] = lo[i]; r[8 + i] = hi[i]; }
  return r;
}

__device__ __forceinline__ v8f wmma_f16(v16h a, v16h b, v8f c) {
  // (neg_a, A, neg_b, B, c_mod, C, reuse_a, reuse_b)
  return __builtin_amdgcn_wmma_f32_16x16x32_f16(false, a, false, b, (short)0, c,
                                                false, false);
}

__device__ __forceinline__ float rowmax16(float v) {
  v = fmaxf(v, __shfl_xor(v, 1, 16));
  v = fmaxf(v, __shfl_xor(v, 2, 16));
  v = fmaxf(v, __shfl_xor(v, 4, 16));
  v = fmaxf(v, __shfl_xor(v, 8, 16));
  return v;
}

__device__ __forceinline__ float rowsum16(float v) {
  v += __shfl_xor(v, 1, 16);
  v += __shfl_xor(v, 2, 16);
  v += __shfl_xor(v, 4, 16);
  v += __shfl_xor(v, 8, 16);
  return v;
}

#if HAVE_TDM
// Issue one TDM load: up-to-2D tile (dim0 contiguous, dim1 rows) from global
// into LDS, with optional LDS-side padding (pad codes per D# spec).
// data_size_code: 0=1B,1=2B,2=4B,3=8B. tile_d1 == 0 -> 1D tile.
// Toolchain uses the 6-arg builtin: (v4u g0, v8i g1, v4i g2, v4i g3, v8i, cpol)
__device__ __forceinline__ void tdm_load_2d(
    unsigned lds_off, const void* gptr, unsigned tensor_d0, unsigned tensor_d1,
    unsigned d0_stride, unsigned tile_d0, unsigned tile_d1,
    unsigned data_size_code, unsigned pad_interval_code,
    unsigned pad_amount_code, bool pad_en) {
  unsigned long long ga = (unsigned long long)(uintptr_t)gptr;
  v4u g0;
  g0[0] = 1u;                                              // count=1, user mode
  g0[1] = lds_off;                                         // lds_addr
  g0[2] = (unsigned)(ga & 0xFFFFFFFFu);                    // global_addr[31:0]
  g0[3] = (unsigned)((ga >> 32) & 0x01FFFFFFu) | (2u << 30);  // addr[56:32] | type=2
  v8i g1;
  unsigned w0 = (data_size_code << 16) | ((pad_en ? 1u : 0u) << 20) |
                (pad_interval_code << 22) | (pad_amount_code << 25);
  g1[0] = (int)w0;                                         // mask=0 | ds | pad cfg
  g1[1] = (int)((tensor_d0 & 0xFFFFu) << 16);              // tensor_dim0[15:0]
  g1[2] = (int)(((tensor_d0 >> 16) & 0xFFFFu) | ((tensor_d1 & 0xFFFFu) << 16));
  g1[3] = (int)(((tensor_d1 >> 16) & 0xFFFFu) | ((tile_d0 & 0xFFFFu) << 16));
  g1[4] = (int)(tile_d1 & 0xFFFFu);                        // tile_dim1 (tile_dim2=0)
  g1[5] = (int)d0_stride;                                  // tensor_dim0_stride lo
  g1[6] = 0;
  g1[7] = 0;
  v4i z4 = {0, 0, 0, 0};
  v8i z8 = {0, 0, 0, 0, 0, 0, 0, 0};
  __builtin_amdgcn_tensor_load_to_lds(g0, g1, z4, z4, z8, 0);
}
#endif

__device__ __forceinline__ unsigned lds_offset_of(const void* p) {
  return (unsigned)(uintptr_t)p;  // low 32 bits of flat shared addr = LDS byte offset
}

// ---------------------------------------------------------------------------
// Kernel 0: convert W (Wq|Wk|Wv, fp32 [E,H]) into f16 transposed W3t[n][k]
// (n = 0..191 output column, k = 0..383), so GEMM B-tiles are k-contiguous.
// ---------------------------------------------------------------------------
__global__ __launch_bounds__(256) void wcvt_kernel(
    const float* __restrict__ Wq, const float* __restrict__ Wk,
    const float* __restrict__ Wv, _Float16* __restrict__ W3t) {
  int idx = blockIdx.x * 256 + threadIdx.x;  // 0 .. 73727
  if (idx >= NCOL * E_) return;
  int n = idx / E_;
  int k = idx - n * E_;
  float w;
  if (n < 64)        w = Wq[k * H_ + n];
  else if (n < 128)  w = Wk[k * H_ + (n - 64)];
  else               w = Wv[k * H_ + (n - 128)];
  W3t[idx] = (_Float16)w;
}

// ---------------------------------------------------------------------------
// Kernel 1: fused QKV GEMM.  C[131072 x 192] = x[131072 x 384] * W3.
// Block = 256 threads (8 waves); each block does 128 rows x all 192 cols.
// W chunks (192 x 32 f16) staged into LDS by the Tensor Data Mover, double
// buffered; TDM padding creates the 40-half row stride (16 DW + 4 DW pad).
// B fragments are software-pipelined one tile ahead so LDS latency hides
// behind XDL WMMA execution.
// ---------------------------------------------------------------------------
__global__ __launch_bounds__(256) void qkv_proj_kernel(
    const float* __restrict__ x, const _Float16* __restrict__ W3t,
    _Float16* __restrict__ Qo, _Float16* __restrict__ Ko,
    _Float16* __restrict__ Vt) {
  __shared__ _Float16 Wt[2][NCOL * WT_STRIDE];  // 2 x 15360 B

  const int tid    = threadIdx.x;
  const int lane   = tid & 31;
  const int wave   = tid >> 5;
  const int half16 = lane >> 4;
  const int l16    = lane & 15;
  const int row0   = blockIdx.x * 128 + wave * 16;

  v8f acc[12];
#pragma unroll
  for (int t = 0; t < 12; ++t) acc[t] = vzero8();

#if HAVE_TDM
  if (wave == 0) {
    // chunk 0 -> buffer 0: 2D tile 32(k) x 192(n) of f16 from W3t (stride 384)
    tdm_load_2d(lds_offset_of(&Wt[0][0]), W3t, E_, NCOL, E_, 32, NCOL,
                /*ds=*/1, /*pad_int: 16DW=*/3, /*pad_amt: 4DW=*/3, true);
  }
#else
  for (int i = tid; i < NCOL * 32; i += 256) {
    int n = i >> 5, kk = i & 31;
    Wt[0][n * WT_STRIDE + kk] = W3t[n * E_ + kk];
  }
#endif

  for (int c = 0; c < 12; ++c) {
    const int k0 = c * 32;
#if HAVE_TDM
    if (wave == 0) {
      if (c + 1 < 12) {
        tdm_load_2d(lds_offset_of(&Wt[(c + 1) & 1][0]), W3t + (k0 + 32), E_,
                    NCOL, E_, 32, NCOL, 1, 3, 3, true);
        __builtin_amdgcn_s_wait_tensorcnt(1);  // chunk c complete (in-order)
      } else {
        __builtin_amdgcn_s_wait_tensorcnt(0);
      }
    }
#else
    if (c + 1 < 12)
      for (int i = tid; i < NCOL * 32; i += 256) {
        int n = i >> 5, kk = i & 31;
        Wt[(c + 1) & 1][n * WT_STRIDE + kk] = W3t[n * E_ + k0 + 32 + kk];
      }
#endif
    __syncthreads();

    // A fragment (16x32 f16): lane<16 -> K{0..7,16..23}, lane>=16 -> +8.
    const float* xrow = x + (size_t)(row0 + l16) * E_ + k0 + half16 * 8;
    __builtin_prefetch(xrow + 32, 0, 1);  // next k-chunk -> global_prefetch_b8
    v4f x0 = *(const v4f*)(xrow);
    v4f x1 = *(const v4f*)(xrow + 4);
    v4f x2 = *(const v4f*)(xrow + 16);
    v4f x3 = *(const v4f*)(xrow + 20);
    v16h a;
#pragma unroll
    for (int i = 0; i < 4; ++i) {
      a[i]      = (_Float16)x0[i];
      a[4 + i]  = (_Float16)x1[i];
      a[8 + i]  = (_Float16)x2[i];
      a[12 + i] = (_Float16)x3[i];
    }

    const _Float16* wb = &Wt[c & 1][0];
    // Software pipeline: B fragment for tile t+1 is loaded before the WMMA on
    // tile t issues, keeping its two ds_load_b128 in flight under the XDL op.
    const _Float16* wp0 = wb + l16 * WT_STRIDE + half16 * 8;
    v16h bcur = cat8(*(const v8h*)wp0, *(const v8h*)(wp0 + 16));
#pragma unroll
    for (int t = 0; t < 12; ++t) {
      v16h bnext = bcur;
      if (t + 1 < 12) {
        const _Float16* wp = wb + ((t + 1) * 16 + l16) * WT_STRIDE + half16 * 8;
        bnext = cat8(*(const v8h*)wp, *(const v8h*)(wp + 16));
      }
      acc[t] = wmma_f16(a, bcur, acc[t]);
      bcur = bnext;
    }
    __syncthreads();  // reads done before TDM reuses this buffer
  }

  // Store D tiles: lane holds rows M = r + 8*half16, col N = l16 of each tile.
  const int rowb = row0 + half16 * 8;
#pragma unroll
  for (int t = 0; t < 12; ++t) {
    int col = t * 16 + l16;
#pragma unroll
    for (int r = 0; r < 8; ++r) {
      int row = rowb + r;
      int bb  = row >> 8;    // / S_
      int ss  = row & 255;   // % S_
      _Float16 vh = (_Float16)acc[t][r];
      if (col < 64)
        Qo[((size_t)bb * S_ + ss) * H_ + col] = vh;
      else if (col < 128)
        Ko[((size_t)bb * S_ + ss) * H_ + (col - 64)] = vh;
      else
        Vt[((size_t)bb * H_ + (col - 128)) * S_ + ss] = vh;
    }
  }
}

// ---------------------------------------------------------------------------
// Kernel 2: causal flash attention, one block per batch (8 waves).
// K[b] staged into LDS via one TDM descriptor (8B elems, pad 4DW per 32DW ->
// 72-half row stride).  Wave w owns q rows [32w, 32w+32), walks key tiles of
// 32; online softmax in f32; P converted C->A layout via per-wave LDS scratch.
// ---------------------------------------------------------------------------
__global__ __launch_bounds__(256) void attn_kernel(
    const _Float16* __restrict__ Qg, const _Float16* __restrict__ Kg,
    const _Float16* __restrict__ Vtg, float* __restrict__ out) {
  __shared__ _Float16 Ks[S_ * KS_STRIDE];      // 36864 B
  __shared__ _Float16 Ps[8 * 16 * PS_STRIDE];  // 10240 B

  const int b      = blockIdx.x;
  const int tid    = threadIdx.x;
  const int lane   = tid & 31;
  const int wave   = tid >> 5;
  const int half16 = lane >> 4;
  const int l16    = lane & 15;

#if HAVE_TDM
  if (wave == 0) {
    // K[b]: 2048 x 8B = 128 rows x 64 halves, 1D with padding every 32 DW.
    tdm_load_2d(lds_offset_of(&Ks[0]), Kg + (size_t)b * S_ * H_,
                /*tensor_d0=*/2048, /*tensor_d1=*/1, /*d0_stride=*/2048,
                /*tile_d0=*/2048, /*tile_d1=*/0,
                /*ds: 8B=*/3, /*pad_int: 32DW=*/4, /*pad_amt: 4DW=*/3, true);
    __builtin_amdgcn_s_wait_tensorcnt(0);
  }
#else
  {
    const _Float16* src = Kg + ((size_t)b * S_ + tid) * H_;
    _Float16* dst = &Ks[tid * KS_STRIDE];
#pragma unroll
    for (int i = 0; i < 8; ++i)
      *(v8h*)(dst + i * 8) = *(const v8h*)(src + i * 8);
  }
#endif

  // Q fragments: 2 M-tiles x 2 H-chunks, A layout (overlaps the K staging).
  v16h aq[2][2];
#pragma unroll
  for (int mt = 0; mt < 2; ++mt) {
    int q = wave * 32 + mt * 16 + l16;
    const _Float16* qrow = Qg + ((size_t)b * S_ + q) * H_;
#pragma unroll
    for (int kt = 0; kt < 2; ++kt) {
      int off = kt * 32 + half16 * 8;
      aq[mt][kt] = cat8(*(const v8h*)(qrow + off), *(const v8h*)(qrow + off + 16));
    }
  }

  __syncthreads();  // K visible to all waves

  float mrow[2][8], lrow[2][8];
  v8f o[2][4];
#pragma unroll
  for (int mt = 0; mt < 2; ++mt) {
#pragma unroll
    for (int r = 0; r < 8; ++r) { mrow[mt][r] = -1e30f; lrow[mt][r] = 0.0f; }
#pragma unroll
    for (int ht = 0; ht < 4; ++ht) o[mt][ht] = vzero8();
  }

  _Float16* Pw = &Ps[wave * 16 * PS_STRIDE];
  const float scale = 0.05103103630798287f;  // 384^-0.5 (embed-dim scaling)

  for (int kt32 = 0; kt32 <= wave; ++kt32) {
    const int kbase = kt32 * 32;

    // Score B fragments (K^T chunks): [kh = key half][hk = h chunk].
    v16h bk[2][2];
#pragma unroll
    for (int kh = 0; kh < 2; ++kh) {
      const _Float16* krow = &Ks[(kbase + kh * 16 + l16) * KS_STRIDE];
#pragma unroll
      for (int hk = 0; hk < 2; ++hk) {
        const _Float16* p = krow + hk * 32 + half16 * 8;
        bk[kh][hk] = cat8(*(const v8h*)p, *(const v8h*)(p + 16));
      }
    }

#pragma unroll
    for (int mt = 0; mt < 2; ++mt) {
      v8f s0 = vzero8(), s1 = vzero8();
      s0 = wmma_f16(aq[mt][0], bk[0][0], s0);
      s0 = wmma_f16(aq[mt][1], bk[0][1], s0);
      s1 = wmma_f16(aq[mt][0], bk[1][0], s1);
      s1 = wmma_f16(aq[mt][1], bk[1][1], s1);

      const int qrowb = wave * 32 + mt * 16 + half16 * 8;
      const int key0  = kbase + l16;
#pragma unroll
      for (int r = 0; r < 8; ++r) {
        int q = qrowb + r;
        float v0 = s0[r] * scale; if (key0 > q)      v0 = -1e30f;
        float v1 = s1[r] * scale; if (key0 + 16 > q) v1 = -1e30f;
        float tmax = rowmax16(fmaxf(v0, v1));
        float mold = mrow[mt][r];
        float mnew = fmaxf(mold, tmax);
        float corr = __expf(mold - mnew);
        float p0   = __expf(v0 - mnew);
        float p1   = __expf(v1 - mnew);
        lrow[mt][r] = lrow[mt][r] * corr + rowsum16(p0 + p1);
        mrow[mt][r] = mnew;
#pragma unroll
        for (int ht = 0; ht < 4; ++ht) o[mt][ht][r] *= corr;
        Pw[(half16 * 8 + r) * PS_STRIDE + l16]      = (_Float16)p0;
        Pw[(half16 * 8 + r) * PS_STRIDE + 16 + l16] = (_Float16)p1;
      }

      // Reload P (16 q x 32 keys) in A layout.
      const _Float16* pr = &Pw[l16 * PS_STRIDE + half16 * 8];
      v16h pa = cat8(*(const v8h*)pr, *(const v8h*)(pr + 16));

      // O += P @ V : B fragments from global Vt[b][h][s] (L2 resident).
#pragma unroll
      for (int ht = 0; ht < 4; ++ht) {
        const _Float16* vp =
            Vtg + ((size_t)b * H_ + ht * 16 + l16) * S_ + kbase + half16 * 8;
        v16h bv = cat8(*(const v8h*)vp, *(const v8h*)(vp + 16));
        o[mt][ht] = wmma_f16(pa, bv, o[mt][ht]);
      }
    }
  }

  // Epilogue: normalize and store fp32 output [B,S,H].
#pragma unroll
  for (int mt = 0; mt < 2; ++mt) {
    int q0 = wave * 32 + mt * 16 + half16 * 8;
#pragma unroll
    for (int ht = 0; ht < 4; ++ht) {
      int h = ht * 16 + l16;
#pragma unroll
      for (int r = 0; r < 8; ++r) {
        out[((size_t)b * S_ + q0 + r) * H_ + h] = o[mt][ht][r] / lrow[mt][r];
      }
    }
  }
}

extern "C" void kernel_launch(void* const* d_in, const int* in_sizes, int n_in,
                              void* d_out, int out_size, void* d_ws, size_t ws_size,
                              hipStream_t stream) {
  (void)in_sizes; (void)n_in; (void)out_size; (void)ws_size;
  const float* x  = (const float*)d_in[0];
  const float* Wq = (const float*)d_in[1];
  const float* Wk = (const float*)d_in[2];
  const float* Wv = (const float*)d_in[3];
  float* out = (float*)d_out;

  const size_t elems = (size_t)B_ * S_ * H_;  // 8,388,608
  _Float16* Q   = (_Float16*)d_ws;            // 16 MB each, f16
  _Float16* K   = Q + elems;
  _Float16* Vt  = K + elems;
  _Float16* W3t = Vt + elems;                 // 144 KB

  wcvt_kernel<<<(NCOL * E_ + 255) / 256, 256, 0, stream>>>(Wq, Wk, Wv, W3t);
  qkv_proj_kernel<<<(B_ * S_) / 128, 256, 0, stream>>>(x, W3t, Q, K, Vt);
  attn_kernel<<<B_, 256, 0, stream>>>(Q, K, Vt, out);
}
